// CoAM_Module_8684423872798
// MI455X (gfx1250) — matrix operand order repaired and checked by reading, not compiled, not run on hardware
//
#include <hip/hip_runtime.h>
#include <hip/hip_bf16.h>
#include <stdint.h>

// Problem dims (from reference): B=16, C=512, H=W=64 -> N=4096
#define B_ 16
#define C_ 512
#define N_ 4096

#define BM 128
#define BN 128
#define BK 64
#define KP 72   // padded LDS k-stride (bf16 elems): 144B rows, 16B aligned, bank-spread

typedef __attribute__((ext_vector_type(16))) __bf16 v16bf;
typedef __attribute__((ext_vector_type(8)))  float  v8f;
typedef unsigned int v4u __attribute__((ext_vector_type(4)));
typedef int          v8i __attribute__((ext_vector_type(8)));
typedef int          v4i __attribute__((ext_vector_type(4)));

#if defined(__has_builtin)
#if __has_builtin(__builtin_amdgcn_tensor_load_to_lds)
#define HAVE_TDM 1
#endif
#endif

#ifdef HAVE_TDM
// ROCm 7.2 clang-22: 5-arg builtin. amdgpu-toolchain (therock headers present): 6-arg.
#if __has_include(<hip/amd_detail/amd_gfx1250_TDM.h>)
#define TDM_CALL(g0, g1, z4a, z4b) \
  __builtin_amdgcn_tensor_load_to_lds((g0), (g1), (z4a), (z4b), (v8i){0,0,0,0,0,0,0,0}, 0)
#else
#define TDM_CALL(g0, g1, z4a, z4b) \
  __builtin_amdgcn_tensor_load_to_lds((g0), (g1), (z4a), (z4b), 0)
#endif
#endif

static __device__ __forceinline__ uint16_t f2bf(float f) {
  union { float f; uint32_t u; } a; a.f = f;
  uint32_t u = a.u;
  return (uint16_t)((u + 0x7FFFu + ((u >> 16) & 1u)) >> 16); // RNE
}

static __device__ __forceinline__ void wait_tensor0() {
#if defined(__has_builtin) && __has_builtin(__builtin_amdgcn_s_wait_tensorcnt)
  __builtin_amdgcn_s_wait_tensorcnt(0);
#else
  asm volatile("s_wait_tensorcnt 0x0" ::: "memory");
#endif
}

#ifdef HAVE_TDM
// TDM 2D tile load: global (bf16, row stride = row_stride elems) -> LDS tile with
// 128B rows padded by 16B (KP=72 layout). Tile = tile_w(=64) x tile_h(=128).
// D# per CDNA5 ISA: group0/group1 bitfields; groups 2/3 unused.
static __device__ __forceinline__ void tdm_load_tile(
    uint32_t lds_byte_off, const uint16_t* gptr,
    uint32_t tensor_w, uint32_t tensor_h, uint32_t row_stride,
    uint32_t tile_w, uint32_t tile_h) {
  uint64_t ga = (uint64_t)(uintptr_t)gptr;
  v4u g0;
  g0.x = 1u;                                   // count=1, is_restore=0, gather=0
  g0.y = lds_byte_off;                         // lds_addr (bytes)
  g0.z = (uint32_t)ga;                         // global_addr[31:0]
  g0.w = (uint32_t)((ga >> 32) & 0x01FFFFFFu)  // global_addr[56:32]
         | (2u << 30);                         // type = 2 ("image")
  v8i g1;
  // data_size=1 (2B), pad_enable, pad_interval=4 (32 DWORDs = 128B),
  // pad_amount=3 (4 DWORDs = 16B)  -> reproduces the KP=72 padded LDS rows.
  g1[0] = (int)((1u << 16) | (1u << 20) | (4u << 22) | (3u << 25));
  g1[1] = (int)((tensor_w & 0xFFFFu) << 16);                                  // dim0 lo
  g1[2] = (int)(((tensor_w >> 16) & 0xFFFFu) | ((tensor_h & 0xFFFFu) << 16)); // dim0 hi | dim1 lo
  g1[3] = (int)(((tensor_h >> 16) & 0xFFFFu) | ((tile_w & 0xFFFFu) << 16));   // dim1 hi | tile_dim0
  g1[4] = (int)(tile_h & 0xFFFFu);             // tile_dim1 | tile_dim2=0
  g1[5] = (int)row_stride;                     // tensor_dim0_stride[31:0]
  g1[6] = 0;                                   // stride0 hi | stride1 lo
  g1[7] = 0;
  v4i z4 = (v4i){0, 0, 0, 0};
  TDM_CALL(g0, g1, z4, z4);
}
#endif

// ---------------------------------------------------------------------------
// Kernel 1: f32 -> bf16 conversion of x1 (v) and x2 (q) into workspace.
// ---------------------------------------------------------------------------
__global__ __launch_bounds__(256) void convert_kernel(
    const float* __restrict__ x1, const float* __restrict__ x2,
    uint16_t* __restrict__ vb, uint16_t* __restrict__ qb) {
  const long i = (long)blockIdx.x * blockDim.x + threadIdx.x; // one float4 each
  const long total4 = (long)B_ * C_ * N_ / 4;
  if (i >= total4) return;
  float4 a = ((const float4*)x1)[i];
  float4 b = ((const float4*)x2)[i];
  uint2 pa, pb;
  pa.x = (uint32_t)f2bf(a.x) | ((uint32_t)f2bf(a.y) << 16);
  pa.y = (uint32_t)f2bf(a.z) | ((uint32_t)f2bf(a.w) << 16);
  pb.x = (uint32_t)f2bf(b.x) | ((uint32_t)f2bf(b.y) << 16);
  pb.y = (uint32_t)f2bf(b.z) | ((uint32_t)f2bf(b.w) << 16);
  ((uint2*)vb)[i] = pa;
  ((uint2*)qb)[i] = pb;
}

// ---------------------------------------------------------------------------
// Fragment loaders from LDS tiles stored as [row][k] with k-stride KP.
// ---------------------------------------------------------------------------
union Frag { uint4 q[2]; v16bf v; };

static __device__ __forceinline__ v16bf load_afrag(const uint16_t* tile, int row, int ks, int lane) {
  int hl = lane >> 4, l16 = lane & 15;
  const uint16_t* p = tile + (row + l16) * KP + ks * 32 + hl * 8;
  Frag u;
  u.q[0] = *(const uint4*)p;
  u.q[1] = *(const uint4*)(p + 16);
  return u.v;
}

static __device__ __forceinline__ v16bf load_bfrag(const uint16_t* tile, int col, int ks, int lane) {
  int hl = lane >> 4, l16 = lane & 15;
  const uint16_t* p = tile + (col + l16) * KP + ks * 32 + hl * 8;
  Frag u;
  u.q[0] = *(const uint4*)p;
  u.q[1] = *(const uint4*)(p + 16);
  return u.v;
}

// One BK=64 chunk of the wave's 32x64 subtile: 2 k-steps x (2x4) WMMAs.
static __device__ __forceinline__ void mma_tile(const uint16_t* At, const uint16_t* Bt,
                                                v8f acc[2][4], int wrow, int wcol, int lane) {
#pragma unroll
  for (int ks = 0; ks < 2; ks++) {
    v16bf af[2], bf[4];
#pragma unroll
    for (int mt = 0; mt < 2; mt++) af[mt] = load_afrag(At, wrow + mt * 16, ks, lane);
#pragma unroll
    for (int nt = 0; nt < 4; nt++) bf[nt] = load_bfrag(Bt, wcol + nt * 16, ks, lane);
#pragma unroll
    for (int mt = 0; mt < 2; mt++)
#pragma unroll
      for (int nt = 0; nt < 4; nt++)
        acc[mt][nt] = __builtin_amdgcn_wmma_f32_16x16x32_bf16(
            false, af[mt], false, bf[nt], (short)0, acc[mt][nt], false, false);
  }
}

// ---------------------------------------------------------------------------
// Kernel 2: energy[b] = q[b] * q[b]^T   (C x C, K = N), bf16 WMMA, f32 out.
// Double-buffered TDM staging: DMA of chunk i+1 overlaps WMMA of chunk i.
// ---------------------------------------------------------------------------
__global__ __launch_bounds__(256) void energy_kernel(
    const uint16_t* __restrict__ qb, float* __restrict__ energy) {
  __shared__ alignas(16) uint16_t As[2][BM * KP];
  __shared__ alignas(16) uint16_t Bs[2][BN * KP];

  const int b  = blockIdx.y;
  const int ti = blockIdx.x / (C_ / BN);
  const int tj = blockIdx.x % (C_ / BN);
  const uint16_t* qbase = qb + (size_t)b * C_ * N_;
  const uint16_t* arow = qbase + (size_t)(ti * BM) * N_;
  const uint16_t* brow = qbase + (size_t)(tj * BN) * N_;

  const int t = threadIdx.x;
  const int lane = t & 31, wave = t >> 5;
  const int wrow = (wave & 3) * 32;   // 0,32,64,96
  const int wcol = (wave >> 2) * 64;  // 0,64

  v8f acc[2][4];
#pragma unroll
  for (int i = 0; i < 2; i++)
#pragma unroll
    for (int j = 0; j < 4; j++) acc[i][j] = (v8f){0,0,0,0,0,0,0,0};

#ifdef HAVE_TDM
  const int NCH = N_ / BK;   // 64 chunks
  if (wave == 0) {
    tdm_load_tile((uint32_t)(uintptr_t)&As[0][0], arow, N_, C_, N_, BK, BM);
    tdm_load_tile((uint32_t)(uintptr_t)&Bs[0][0], brow, N_, C_, N_, BK, BN);
    wait_tensor0();
  }
  __syncthreads();
  for (int i = 0; i < NCH; i++) {
    const int cur = i & 1;
    if (wave == 0 && i + 1 < NCH) {  // kick next chunk's DMA before computing
      tdm_load_tile((uint32_t)(uintptr_t)&As[cur ^ 1][0], arow + (i + 1) * BK,
                    N_, C_, N_, BK, BM);
      tdm_load_tile((uint32_t)(uintptr_t)&Bs[cur ^ 1][0], brow + (i + 1) * BK,
                    N_, C_, N_, BK, BN);
    }
    mma_tile(As[cur], Bs[cur], acc, wrow, wcol, lane);
    if (wave == 0 && i + 1 < NCH) wait_tensor0();
    __syncthreads();
  }
#else
  for (int k0 = 0; k0 < N_; k0 += BK) {
#pragma unroll
    for (int r = 0; r < 4; r++) {
      int idx = t + r * 256;          // 0..1023 uint4 units
      int row = idx >> 3;
      int c16 = idx & 7;
      uint4 ga = *(const uint4*)(arow + (size_t)row * N_ + k0 + c16 * 8);
      *(uint4*)(&As[0][row * KP + c16 * 8]) = ga;
      uint4 gb = *(const uint4*)(brow + (size_t)row * N_ + k0 + c16 * 8);
      *(uint4*)(&Bs[0][row * KP + c16 * 8]) = gb;
    }
    __syncthreads();
    mma_tile(As[0], Bs[0], acc, wrow, wcol, lane);
    __syncthreads();
  }
#endif

  // Store f32 energy tile. C/D layout: lane l -> n = l%16; VGPR r -> m = r + 8*(l/16).
  float* ebase = energy + ((size_t)b * C_ + ti * BM) * C_ + tj * BN;
  const int hl = lane >> 4, l16 = lane & 15;
#pragma unroll
  for (int mt = 0; mt < 2; mt++)
#pragma unroll
    for (int nt = 0; nt < 4; nt++)
#pragma unroll
      for (int r = 0; r < 8; r++)
        ebase[(size_t)(wrow + mt * 16 + r + 8 * hl) * C_ + (wcol + nt * 16 + l16)] =
            acc[mt][nt][r];
}

// ---------------------------------------------------------------------------
// Kernel 3: row softmax of (max - energy): attn_j = exp(min_e - e_j)/sum.
// One wave per row of 512; 8 waves/block. Output bf16 directly.
// ---------------------------------------------------------------------------
__global__ __launch_bounds__(256) void softmax_kernel(
    const float* __restrict__ energy, uint16_t* __restrict__ attn) {
  const int row  = blockIdx.x * 8 + (threadIdx.x >> 5);
  const int lane = threadIdx.x & 31;
  const float* e = energy + (size_t)row * C_;

  float vals[16];
  float mn = 3.4e38f;
#pragma unroll
  for (int i = 0; i < 16; i++) { vals[i] = e[lane + i * 32]; mn = fminf(mn, vals[i]); }
#pragma unroll
  for (int off = 16; off; off >>= 1) mn = fminf(mn, __shfl_xor(mn, off, 32));

  float p[16], sum = 0.f;
#pragma unroll
  for (int i = 0; i < 16; i++) { p[i] = __expf(mn - vals[i]); sum += p[i]; }
#pragma unroll
  for (int off = 16; off; off >>= 1) sum += __shfl_xor(sum, off, 32);

  const float inv = 1.0f / sum;
  uint16_t* o = attn + (size_t)row * C_;
#pragma unroll
  for (int i = 0; i < 16; i++) o[lane + i * 32] = f2bf(p[i] * inv);
}

// ---------------------------------------------------------------------------
// av_kernel helpers: software-pipelined v -> v^T staging (global->VGPR prefetch,
// DS interleave store after compute).
// ---------------------------------------------------------------------------
static __device__ __forceinline__ void load_b_regs(uint4 pre[4], const uint16_t* vbase,
                                                   int k0, int tn, int t) {
#pragma unroll
  for (int u = 0; u < 2; u++) {
    int unit = t + u * 256;       // 0..511 : 32 k-pairs x 16 n-octets
    int kp = unit >> 4;
    int no = unit & 15;
    const uint16_t* g0 = vbase + (size_t)(k0 + 2 * kp) * N_ + tn * BN + no * 8;
    pre[2 * u]     = *(const uint4*)g0;
    pre[2 * u + 1] = *(const uint4*)(g0 + N_);
  }
}

static __device__ __forceinline__ void store_b_lds(uint16_t* Bt, const uint4 pre[4], int t) {
#pragma unroll
  for (int u = 0; u < 2; u++) {
    int unit = t + u * 256;
    int kp = unit >> 4;
    int no = unit & 15;
    union { uint4 q; uint16_t h[8]; } r0, r1;
    r0.q = pre[2 * u];
    r1.q = pre[2 * u + 1];
#pragma unroll
    for (int d = 0; d < 8; d++) {
      uint32_t dw = (uint32_t)r0.h[d] | ((uint32_t)r1.h[d] << 16);
      *(uint32_t*)&Bt[(no * 8 + d) * KP + 2 * kp] = dw;
    }
  }
}

// ---------------------------------------------------------------------------
// Kernel 4: out = gamma * (attn * v) + x1.  (C x N, K = C) bf16 WMMA.
// A tile: double-buffered TDM. B tile: pipelined manual transpose staging.
// ---------------------------------------------------------------------------
__global__ __launch_bounds__(256) void av_kernel(
    const uint16_t* __restrict__ attn, const uint16_t* __restrict__ vb,
    const float* __restrict__ x1, const float* __restrict__ gamma,
    float* __restrict__ out) {
  __shared__ alignas(16) uint16_t As[2][BM * KP]; // attn tile [i][j]
  __shared__ alignas(16) uint16_t Bs[2][BN * KP]; // v^T tile  [n][j]

  const int b  = blockIdx.z;
  const int ti = blockIdx.y;
  const int tn = blockIdx.x;
  const uint16_t* abase = attn + (size_t)b * C_ * C_;
  const uint16_t* arow  = abase + (size_t)(ti * BM) * C_;
  const uint16_t* vbase = vb + (size_t)b * C_ * N_;

  const int t = threadIdx.x;
  const int lane = t & 31, wave = t >> 5;
  const int wrow = (wave & 3) * 32;
  const int wcol = (wave >> 2) * 64;

  v8f acc[2][4];
#pragma unroll
  for (int i = 0; i < 2; i++)
#pragma unroll
    for (int j = 0; j < 4; j++) acc[i][j] = (v8f){0,0,0,0,0,0,0,0};

#ifdef HAVE_TDM
  const int NCH = C_ / BK;   // 8 chunks
  uint4 pre[4];
  load_b_regs(pre, vbase, 0, tn, t);
  if (wave == 0)
    tdm_load_tile((uint32_t)(uintptr_t)&As[0][0], arow, C_, C_, C_, BK, BM);
  store_b_lds(Bs[0], pre, t);
  if (wave == 0) wait_tensor0();
  __syncthreads();
  for (int i = 0; i < NCH; i++) {
    const int cur = i & 1;
    if (i + 1 < NCH) {
      load_b_regs(pre, vbase, (i + 1) * BK, tn, t);   // prefetch next B chunk
      if (wave == 0)
        tdm_load_tile((uint32_t)(uintptr_t)&As[cur ^ 1][0], arow + (i + 1) * BK,
                      C_, C_, C_, BK, BM);
    }
    mma_tile(As[cur], Bs[cur], acc, wrow, wcol, lane);
    if (i + 1 < NCH) {
      store_b_lds(Bs[cur ^ 1], pre, t);
      if (wave == 0) wait_tensor0();
    }
    __syncthreads();
  }
#else
  for (int k0 = 0; k0 < C_; k0 += BK) {
#pragma unroll
    for (int r = 0; r < 4; r++) {
      int idx = t + r * 256;
      int row = idx >> 3;
      int c16 = idx & 7;
      uint4 ga = *(const uint4*)(arow + (size_t)row * C_ + k0 + c16 * 8);
      *(uint4*)(&As[0][row * KP + c16 * 8]) = ga;
    }
    uint4 pre[4];
    load_b_regs(pre, vbase, k0, tn, t);
    store_b_lds(Bs[0], pre, t);
    __syncthreads();
    mma_tile(As[0], Bs[0], acc, wrow, wcol, lane);
    __syncthreads();
  }
#endif

  // Epilogue: out = gamma*acc + x1 (residual), fused.
  const float g = gamma[0];
  const int hl = lane >> 4, l16 = lane & 15;
  const size_t obase = (size_t)b * C_ * N_ + (size_t)(ti * BM) * N_ + tn * BN;
#pragma unroll
  for (int mt = 0; mt < 2; mt++)
#pragma unroll
    for (int nt = 0; nt < 4; nt++)
#pragma unroll
      for (int r = 0; r < 8; r++) {
        size_t idx = obase + (size_t)(wrow + mt * 16 + r + 8 * hl) * N_ +
                     (wcol + nt * 16 + l16);
        out[idx] = g * acc[mt][nt][r] + x1[idx];
      }
}

// ---------------------------------------------------------------------------
extern "C" void kernel_launch(void* const* d_in, const int* in_sizes, int n_in,
                              void* d_out, int out_size, void* d_ws, size_t ws_size,
                              hipStream_t stream) {
  const float* x1    = (const float*)d_in[0];
  const float* x2    = (const float*)d_in[1];
  const float* gamma = (const float*)d_in[2];
  float* out = (float*)d_out;

  char* ws = (char*)d_ws;
  const size_t tensor_bf16 = (size_t)B_ * C_ * N_ * sizeof(uint16_t); // 64 MiB
  const size_t energy_f32  = (size_t)B_ * C_ * C_ * sizeof(float);    // 16 MiB
  uint16_t* qb     = (uint16_t*)ws;
  uint16_t* vbuf   = (uint16_t*)(ws + tensor_bf16);
  float*    energy = (float*)   (ws + 2 * tensor_bf16);
  uint16_t* attn   = (uint16_t*)(ws + 2 * tensor_bf16 + energy_f32);

  // 1) f32 -> bf16 staging of both operands.
  const long total4 = (long)B_ * C_ * N_ / 4;
  convert_kernel<<<dim3((unsigned)((total4 + 255) / 256)), dim3(256), 0, stream>>>(
      x1, x2, vbuf, qb);

  // 2) energy = q q^T  (bf16 WMMA, double-buffered TDM tiles).
  energy_kernel<<<dim3((C_ / BM) * (C_ / BN), B_), dim3(256), 0, stream>>>(qb, energy);

  // 3) softmax(max - energy) == exp(min - e)/sum, emitted as bf16.
  softmax_kernel<<<dim3(B_ * C_ / 8), dim3(256), 0, stream>>>(energy, attn);

  // 4) out = gamma * (attn v) + x1  (bf16 WMMA, fused residual epilogue).
  av_kernel<<<dim3(N_ / BN, C_ / BM, B_), dim3(256), 0, stream>>>(
      attn, vbuf, x1, gamma, out);
}
